// PrePolicyRNN_61280593379726
// MI455X (gfx1250) — matrix operand order repaired and checked by reading, not compiled, not run on hardware
//
#include <hip/hip_runtime.h>
#include <hip/hip_bf16.h>
#include <stdint.h>

// ---------------------------------------------------------------------------
// PrePolicyRNN for MI455X (gfx1250): bf16 WMMA GEMMs + persistent cluster-
// synced GRU recurrence with TDM/async LDS staging.
// Shapes: T=512 B=64 OBS=512 H=1024 OUT=256.
// ---------------------------------------------------------------------------

#define T_   512
#define B_   64
#define OBS_ 512
#define H_   1024
#define H3_  3072
#define OUT_ 256
#define TB_  (T_ * B_)   // 32768

// ---- CDNA5 feature probes (compile-safe: fall back if builtin absent) ----
#if __has_builtin(__builtin_amdgcn_tensor_load_to_lds)
#define HAVE_TDM 1
#else
#define HAVE_TDM 0
#endif
#if __has_builtin(__builtin_amdgcn_global_load_async_to_lds_b128)
#define HAVE_ASYNC 1
#else
#define HAVE_ASYNC 0
#endif
#if __has_builtin(__builtin_amdgcn_s_wait_asynccnt)
#define WAIT_ASYNC0() __builtin_amdgcn_s_wait_asynccnt(0)
#else
#define WAIT_ASYNC0() asm volatile("s_wait_asynccnt 0x0" ::: "memory")
#endif

typedef __attribute__((ext_vector_type(16))) __bf16 v16bf;
typedef __attribute__((ext_vector_type(8)))  float  v8f;
typedef __attribute__((ext_vector_type(4)))  unsigned int v4u;
typedef __attribute__((ext_vector_type(8)))  int    v8i;
typedef __attribute__((ext_vector_type(4)))  int    v4i;

#if HAVE_ASYNC
// async-LDS builtin takes v4i* in AS1 (global) and AS3 (LDS)
typedef __attribute__((address_space(1))) v4i* gv4ip;
typedef __attribute__((address_space(3))) v4i* lv4ip;
#endif

union FragB16 { v16bf v; uint4 q[2]; };

static __device__ __forceinline__ v8f vzero8() {
    v8f z;
    for (int i = 0; i < 8; ++i) z[i] = 0.f;
    return z;
}

// Branch-free gate activations on v_exp_f32 / v_rcp_f32.
static __device__ __forceinline__ float fast_sigmoid(float x) {
    return __builtin_amdgcn_rcpf(1.f + __expf(-x));
}
static __device__ __forceinline__ float fast_tanh(float x) {
    return 1.f - 2.f * __builtin_amdgcn_rcpf(1.f + __expf(2.f * x));
}

// ---------------------------------------------------------------- prep kernels
__global__ void cvt_f32_bf16_kernel(const float* __restrict__ src,
                                    __bf16* __restrict__ dst, long n) {
    long i = (long)blockIdx.x * blockDim.x + threadIdx.x;
    if (i < n) dst[i] = (__bf16)src[i];
}

// src: f32 [R][C] row-major  ->  dst: bf16 [C][dstLd] (transposed)
__global__ void transpose_bf16_kernel(const float* __restrict__ src,
                                      __bf16* __restrict__ dst,
                                      int R, int C, int dstLd) {
    long i = (long)blockIdx.x * blockDim.x + threadIdx.x;
    if (i < (long)R * C) {
        int r = (int)(i / C), c = (int)(i % C);
        dst[(size_t)c * dstLd + r] = (__bf16)src[i];
    }
}

// h0 = dones[0] ? 0 : hidden  (bf16); also reset the global barrier counter.
__global__ void init_h_kernel(const float* __restrict__ hidden,
                              const unsigned char* __restrict__ dones,
                              __bf16* __restrict__ h0,
                              unsigned int* __restrict__ bar) {
    int i = blockIdx.x * blockDim.x + threadIdx.x;
    if (i < B_ * H_) {
        int b = i >> 10;
        h0[i] = dones[b] ? (__bf16)0.f : (__bf16)hidden[i];
    }
    if (blockIdx.x == 0 && threadIdx.x == 0) *bar = 0u;
}

// --------------------------------------------------------------- batched GEMM
// C[M x N] = A(bf16,row-major,lda) @ Bt(bf16,[N][ldb])^T  + bias, f32 accum.
// Block: 256 thr (8 waves), WG tile 64(M) x 128(N); wave tile 64 x 16.
// A-tile staging is double-buffered via async global->LDS DMA: one barrier
// per K-step, next tile's DMA issued before the WMMA stream on the current.
// MODE 0: relu, store bf16 | MODE 1: store f32 | MODE 2: +obs_last*w_last, f32
#define SA_STRIDE 2560  // 64 rows x 40 bf16 (padded, conflict-free)

#if HAVE_ASYNC
#define STAGE_A(bufsel, kk)                                                  \
    __builtin_amdgcn_global_load_async_to_lds_b128(                          \
        (gv4ip)(aptr + (kk)), (lv4ip)(sdst + (bufsel) * SA_STRIDE), 0, 0)
#else
#define STAGE_A(bufsel, kk)                                                  \
    (*(uint4*)(sdst + (bufsel) * SA_STRIDE) = *(const uint4*)(aptr + (kk)))
#endif

template <int MODE>
__global__ __launch_bounds__(256, 2) void gemm_bf16_kernel(
    const __bf16* __restrict__ A, int lda,
    const __bf16* __restrict__ Bt, int ldb,
    const float* __restrict__ bias,
    void* __restrict__ Cout, int ldc, int K,
    const float* __restrict__ obsf) {
    __shared__ __bf16 sA[2 * SA_STRIDE];  // ping-pong 64x32 K-slabs

    const int tid = threadIdx.x;
    const int w = tid >> 5, l = tid & 31;
    const int lo = l & 15, hi = l >> 4;
    const int m0 = blockIdx.x * 64;
    const int n0 = blockIdx.y * 128 + w * 16;

    v8f acc[4];
    for (int i = 0; i < 4; ++i) acc[i] = vzero8();

    const int srow = tid >> 2;            // 0..63
    const int scol = (tid & 3) * 8;       // 0,8,16,24
    const __bf16* aptr = A + (size_t)(m0 + srow) * lda + scol;
    __bf16* sdst = &sA[srow * 40 + scol];
    const __bf16* bptr = Bt + (size_t)(n0 + lo) * ldb + hi * 8;

    STAGE_A(0, 0);  // prologue: DMA first K-slab

    int buf = 0;
    for (int k0 = 0; k0 < K; k0 += 32, buf ^= 1) {
        WAIT_ASYNC0();       // this wave's DMA into sA[buf] complete
        __syncthreads();     // all waves' DMA visible; prev reads retired
        if (k0 + 32 < K) STAGE_A(buf ^ 1, k0 + 32);

        FragB16 b;
        b.q[0] = *(const uint4*)(bptr + k0);
        b.q[1] = *(const uint4*)(bptr + k0 + 16);
        if (k0 + 32 < K) __builtin_prefetch(bptr + k0 + 32, 0, 3);

        const __bf16* abase = &sA[buf * SA_STRIDE];
#pragma unroll
        for (int mt = 0; mt < 4; ++mt) {
            FragB16 a;
            const __bf16* ap = abase + (mt * 16 + lo) * 40 + hi * 8;
            a.q[0] = *(const uint4*)(ap);
            a.q[1] = *(const uint4*)(ap + 16);
            acc[mt] = __builtin_amdgcn_wmma_f32_16x16x32_bf16(
                false, a.v, false, b.v, (short)0, acc[mt], false, false);
        }
    }

    const int n = n0 + lo;
    const float bn = bias[n];
    float wl = 0.f;
    if (MODE == 2) wl = (float)Bt[(size_t)n * ldb + 1024];

#pragma unroll
    for (int mt = 0; mt < 4; ++mt) {
        const int mbase = m0 + mt * 16 + hi * 8;
#pragma unroll
        for (int r = 0; r < 8; ++r) {
            const int m = mbase + r;
            float v = acc[mt][r] + bn;
            if (MODE == 0) {
                v = v > 0.f ? v : 0.f;
                ((__bf16*)Cout)[(size_t)m * ldc + n] = (__bf16)v;
            } else if (MODE == 1) {
                ((float*)Cout)[(size_t)m * ldc + n] = v;
            } else {
                v += obsf[(size_t)m * OBS_ + (OBS_ - 1)] * wl;
                ((float*)Cout)[(size_t)m * ldc + n] = v;
            }
        }
    }
}

// ------------------------------------------------------- persistent GRU scan
// LDS h layout: pad 8B after every 1024B (TDM pad_interval=1024B, amount=8B),
// i.e. element (row,c) lives at row*1032 + c + (c>=512 ? 4 : 0).  Row stride
// 2064B = 516 dwords == 4 (mod 64 banks) -> conflict-free WMMA A fragments.
#define HP_ 1032
static __device__ __forceinline__ int hidx(int row, int c) {
    return row * HP_ + c + ((c >= 512) ? 4 : 0);
}

__global__ __launch_bounds__(256, 1) void rnn_kernel(
    const __bf16* __restrict__ WhT,   // [3*H][H]
    const float* __restrict__ gi,     // [T*B][3H]  precomputed input gates
    const float* __restrict__ b_hn,   // [H]
    const unsigned char* __restrict__ dones,  // [T*B]
    __bf16* __restrict__ h0, __bf16* __restrict__ h1,
    __bf16* __restrict__ ys,          // [T*B][H] bf16 (feeds output GEMM)
    float* __restrict__ hid_out,      // [B][H] f32 (final hidden -> d_out)
    unsigned int* __restrict__ bar) {
    extern __shared__ __bf16 sh[];    // 64 x 1032

    const int tid = threadIdx.x;
    const int w = tid >> 5, l = tid & 31;
    const int lo = l & 15, hi = l >> 4;
    const int wm = w >> 2;            // 0..1 -> 32 batch rows
    const int wn = w & 3;             // 0..3 -> 16 h-columns
    const int ncol = blockIdx.x * 64 + wn * 16 + lo;
    const unsigned nwg = gridDim.x;

    const __bf16* bb[3];
    for (int g = 0; g < 3; ++g)
        bb[g] = WhT + (size_t)(g * H_ + ncol) * H_ + hi * 8;

#pragma unroll 1
    for (int t = 0; t < T_; ++t) {
        const __bf16* src = (t & 1) ? h1 : h0;
        __bf16* dst = (t & 1) ? h0 : h1;

        // ---- stage h_t into LDS ----
#if HAVE_TDM
        if (w == 0) {
            // Tensor DMA: 2D tile 1024x64 bf16, LDS padding 2 dwords / 256.
            unsigned ldsa = (unsigned)(uintptr_t)sh;   // low 32b = LDS offset
            unsigned long long ga = (unsigned long long)(uintptr_t)src;
            v4u g0;
            g0[0] = 1u;                                   // count=1, load, user
            g0[1] = ldsa;                                 // lds_addr
            g0[2] = (unsigned)(ga & 0xffffffffu);         // global_addr[31:0]
            g0[3] = (unsigned)((ga >> 32) & 0x1ffffffu)   // global_addr[56:32]
                    | (2u << 30);                         // type=2 (image)
            v8i g1;
            g1[0] = (1 << 16)      // data_size = 2B
                  | (1 << 20)      // pad_enable
                  | (7 << 22)      // pad_interval = 256 dwords (1024B)
                  | (1 << 25);     // pad_amount = 2 dwords (8B)
            g1[1] = (int)(1024u << 16);   // tensor_dim0 = 1024 (elems)
            g1[2] = (int)(64u << 16);     // tensor_dim1 = 64
            g1[3] = (int)(1024u << 16);   // tile_dim0 = 1024
            g1[4] = 64;                   // tile_dim1 = 64, tile_dim2 = 0
            g1[5] = 1024;                 // tensor_dim0_stride = 1024 elems
            g1[6] = 0;
            g1[7] = 0;
            v4i gz4 = {0, 0, 0, 0};
            v8i gz8 = {0, 0, 0, 0, 0, 0, 0, 0};
            __builtin_amdgcn_tensor_load_to_lds(g0, g1, gz4, gz4, gz8, 0);
            __builtin_amdgcn_s_wait_tensorcnt(0);
        }
        __syncthreads();
#else
        for (int i = tid; i < (B_ * H_ / 8); i += 256) {
            int row = i >> 7;
            int c = (i & 127) * 8;
#if HAVE_ASYNC
            __builtin_amdgcn_global_load_async_to_lds_b128(
                (gv4ip)(src + row * H_ + c), (lv4ip)&sh[hidx(row, c)], 0, 0);
#else
            *(uint4*)&sh[hidx(row, c)] = *(const uint4*)(src + row * H_ + c);
#endif
        }
#if HAVE_ASYNC
        WAIT_ASYNC0();
#endif
        __syncthreads();
#endif

        v8f acc[3][2];
        for (int g = 0; g < 3; ++g)
            for (int mt = 0; mt < 2; ++mt) acc[g][mt] = vzero8();

        // ---- gh = h @ Wh for this wave's (r,z,n) column triplet ----
#pragma unroll 1
        for (int k0 = 0; k0 < H_; k0 += 32) {
            const int kadj = (k0 >= 512) ? 4 : 0;
            FragB16 a[2];
#pragma unroll
            for (int mt = 0; mt < 2; ++mt) {
                const __bf16* ap =
                    &sh[(wm * 32 + mt * 16 + lo) * HP_ + k0 + kadj + hi * 8];
                a[mt].q[0] = *(const uint4*)ap;
                a[mt].q[1] = *(const uint4*)(ap + 16);
            }
#pragma unroll
            for (int g = 0; g < 3; ++g) {
                FragB16 b;
                b.q[0] = *(const uint4*)(bb[g] + k0);
                b.q[1] = *(const uint4*)(bb[g] + k0 + 16);
                if (k0 + 32 < H_) __builtin_prefetch(bb[g] + k0 + 32, 0, 3);
#pragma unroll
                for (int mt = 0; mt < 2; ++mt)
                    acc[g][mt] = __builtin_amdgcn_wmma_f32_16x16x32_bf16(
                        false, a[mt].v, false, b.v, (short)0, acc[g][mt],
                        false, false);
            }
        }

        // ---- fused GRU gate epilogue (branch-free activations) ----
        const float bhn = b_hn[ncol];
#pragma unroll
        for (int mt = 0; mt < 2; ++mt) {
            const int mb = wm * 32 + mt * 16 + hi * 8;
#pragma unroll
            for (int r = 0; r < 8; ++r) {
                const int m = mb + r;
                const float* gp = gi + ((size_t)t * B_ + m) * H3_;
                float rr = fast_sigmoid(gp[ncol] + acc[0][mt][r]);
                float zz = fast_sigmoid(gp[H_ + ncol] + acc[1][mt][r]);
                float nn = fast_tanh(gp[2 * H_ + ncol] +
                                     rr * (acc[2][mt][r] + bhn));
                float hold = (float)sh[hidx(m, ncol)];
                float hnew = (1.f - zz) * nn + zz * hold;
                ys[((size_t)t * B_ + m) * H_ + ncol] = (__bf16)hnew;
                float hstore = hnew;
                if (t < T_ - 1 && dones[(t + 1) * B_ + m]) hstore = 0.f;
                dst[m * H_ + ncol] = (__bf16)hstore;
                if (t == T_ - 1) hid_out[m * H_ + ncol] = hnew;
            }
        }

        // ---- step barrier: release stores, sync 16 WGs, invalidate WGP$ ----
        __threadfence();
        __syncthreads();
        __builtin_amdgcn_s_cluster_barrier();  // NOP if dispatch not clustered
        if (tid == 0) {
            __hip_atomic_fetch_add(bar, 1u, __ATOMIC_RELEASE,
                                   __HIP_MEMORY_SCOPE_AGENT);
            const unsigned tgt = nwg * (unsigned)(t + 1);
            while (__hip_atomic_load(bar, __ATOMIC_ACQUIRE,
                                     __HIP_MEMORY_SCOPE_AGENT) < tgt)
                __builtin_amdgcn_s_sleep(2);
        }
        __syncthreads();
        asm volatile("global_inv scope:SCOPE_DEV\n\t"
                     "s_wait_loadcnt 0x0" ::: "memory");
    }
}

// ------------------------------------------------------------------- launcher
extern "C" void kernel_launch(void* const* d_in, const int* in_sizes, int n_in,
                              void* d_out, int out_size, void* d_ws,
                              size_t ws_size, hipStream_t stream) {
    const float* hidden = (const float*)d_in[0];
    const float* obs = (const float*)d_in[1];
    const unsigned char* dones = (const unsigned char*)d_in[2];  // jax bool: 1B
    const float* W_emb = (const float*)d_in[3];
    const float* b_emb = (const float*)d_in[4];
    const float* Wi = (const float*)d_in[5];
    const float* bi = (const float*)d_in[6];
    const float* Wh = (const float*)d_in[7];
    const float* b_hn = (const float*)d_in[8];
    const float* W_out = (const float*)d_in[9];
    const float* b_out = (const float*)d_in[10];
    float* out = (float*)d_out;  // [B*H hidden | T*B*OUT pre_policy]

    char* ws = (char*)d_ws;
    auto alloc = [&](size_t bytes) -> char* {
        char* p = ws;
        ws += (bytes + 255) & ~(size_t)255;
        return p;
    };
    __bf16* obsb  = (__bf16*)alloc((size_t)TB_ * OBS_ * 2);
    __bf16* embb  = (__bf16*)alloc((size_t)TB_ * H_ * 2);
    float*  gibuf = (float*) alloc((size_t)TB_ * H3_ * 4);
    __bf16* ysb   = (__bf16*)alloc((size_t)TB_ * H_ * 2);
    __bf16* WembT = (__bf16*)alloc((size_t)H_ * OBS_ * 2);
    __bf16* WiT   = (__bf16*)alloc((size_t)H3_ * H_ * 2);
    __bf16* WhT   = (__bf16*)alloc((size_t)H3_ * H_ * 2);
    __bf16* WoutT = (__bf16*)alloc((size_t)OUT_ * 1040 * 2);  // K 1025->1040
    __bf16* hb0   = (__bf16*)alloc((size_t)B_ * H_ * 2);
    __bf16* hb1   = (__bf16*)alloc((size_t)B_ * H_ * 2);
    unsigned int* bar = (unsigned int*)alloc(256);

    cvt_f32_bf16_kernel<<<(TB_ * OBS_) / 256, 256, 0, stream>>>(
        obs, obsb, (long)TB_ * OBS_);
    transpose_bf16_kernel<<<(OBS_ * H_) / 256, 256, 0, stream>>>(
        W_emb, WembT, OBS_, H_, OBS_);
    transpose_bf16_kernel<<<(H_ * H3_) / 256, 256, 0, stream>>>(
        Wi, WiT, H_, H3_, H_);
    transpose_bf16_kernel<<<(H_ * H3_) / 256, 256, 0, stream>>>(
        Wh, WhT, H_, H3_, H_);
    transpose_bf16_kernel<<<(1025 * OUT_) / 256, 256, 0, stream>>>(
        W_out, WoutT, 1025, OUT_, 1040);
    init_h_kernel<<<(B_ * H_) / 256, 256, 0, stream>>>(hidden, dones, hb0, bar);

    // Stage A: emb = relu(obs @ W_emb + b_emb) -> bf16
    gemm_bf16_kernel<0><<<dim3(TB_ / 64, H_ / 128), 256, 0, stream>>>(
        obsb, OBS_, WembT, OBS_, b_emb, embb, H_, OBS_, nullptr);
    // Stage B: gi = emb @ Wi + bi -> f32 (hoisted out of the scan)
    gemm_bf16_kernel<1><<<dim3(TB_ / 64, H3_ / 128), 256, 0, stream>>>(
        embb, H_, WiT, H_, bi, gibuf, H3_, H_, nullptr);

    // Stage C: persistent GRU recurrence (16 WGs, 132 KB dynamic LDS each)
    const int rnn_lds = 64 * HP_ * 2;
    (void)hipFuncSetAttribute((const void*)rnn_kernel,
                              hipFuncAttributeMaxDynamicSharedMemorySize,
                              rnn_lds);
    rnn_kernel<<<16, 256, rnn_lds, stream>>>(WhT, gibuf, b_hn, dones, hb0, hb1,
                                             ysb, out, bar);

    // Stage D: pre = ys @ W_out[0:1024] + obs_last*w_last + b_out -> f32
    gemm_bf16_kernel<2><<<dim3(TB_ / 64, OUT_ / 128), 256, 0, stream>>>(
        ysb, H_, WoutT, 1040, b_out, out + B_ * H_, OUT_, H_, obs);
}
